// Gemma3Attention_60859686584670
// MI455X (gfx1250) — matrix operand order repaired and checked by reading, not compiled, not run on hardware
//
#include <hip/hip_runtime.h>
#include <hip/hip_bf16.h>
#include <stdint.h>

typedef _Float16 half_t;
typedef half_t v16h __attribute__((ext_vector_type(16)));
typedef half_t v8h  __attribute__((ext_vector_type(8)));
typedef float  v8f  __attribute__((ext_vector_type(8)));
typedef uint32_t v4u __attribute__((ext_vector_type(4)));
typedef int      v8i __attribute__((ext_vector_type(8)));
typedef int      v4i __attribute__((ext_vector_type(4)));

#define S_LEN 2048
#define HIDSZ 3072
#define NHEAD 24
#define NKVH  8
#define HDIM  128
#define KVD   1024
#define SCALING 0.08838834764831845f
#define SOFTCAP 50.0f
#define NEGBIG  (-1000000000.0f)
#define RMSEPS  1e-6f

static __device__ inline v8f wmma16(v16h a, v16h b, v8f c) {
  return __builtin_amdgcn_wmma_f32_16x16x32_f16(false, a, false, b, (short)0, c, false, false);
}

static __device__ inline v16h cat8(v8h lo, v8h hi) {
  return __builtin_shufflevector(lo, hi, 0,1,2,3,4,5,6,7,8,9,10,11,12,13,14,15);
}

// A fragment (16x32 f16): lane holds row M=lane%16; K chunks [k0+g*8,+8) and [k0+16+g*8,+8)
static __device__ inline v16h load_a_frag(const half_t* base, int ld, int row, int k0, int g) {
  const half_t* p = base + (size_t)row * ld + k0 + g * 8;
  return cat8(*(const v8h*)p, *(const v8h*)(p + 16));
}

// B fragment (32x16 f16): lane holds col N=lane%16; K chunks [k0+g*8,+8) and [k0+16+g*8,+8)
static __device__ inline v16h load_b_frag(const half_t* base, int ld, int col, int k0, int g) {
  const half_t* p = base + (size_t)col * ld + k0 + g * 8;
  return cat8(*(const v8h*)p, *(const v8h*)(p + 16));
}

// ---- Tensor Data Mover: 2D tile (tile_dim0=128 halves contiguous, tile_dim1=32 rows)
// from a [2048 x 128] f16 tensor (row pitch 128 elements) into LDS at byte offset lds_off.
// D# packing per CDNA5 ISA.
static __device__ inline void tdm_load_k_tile(uint64_t gaddr, uint32_t lds_off) {
  v4u g0;
  g0[0] = 1u;                                            // count=1, user desc, no gather
  g0[1] = lds_off;                                       // lds_addr [63:32]
  g0[2] = (uint32_t)gaddr;                               // global_addr[31:0]
  g0[3] = (uint32_t)((gaddr >> 32) & 0x01FFFFFFu)        // global_addr[56:32]
        | (2u << 30);                                    // type=2 ("image")
  v8i g1;
  g1[0] = 0x00010000;        // workgroup_mask=0, data_size=1 (2-byte elems)
  g1[1] = (int)(128u << 16); // tensor_dim0 low16=128 at [31:16]
  g1[2] = (int)(2048u << 16);// tensor_dim0 hi16=0 | tensor_dim1 low16=2048
  g1[3] = (int)(128u << 16); // tensor_dim1 hi16=0 | tile_dim0=128
  g1[4] = 32;                // tile_dim1=32 | tile_dim2=0
  g1[5] = 128;               // tensor_dim0_stride low32 = 128
  g1[6] = 0;                 // stride0 hi16 | tensor_dim1_stride low16 = 0
  g1[7] = 0;
  v4i z4 = {0,0,0,0};
#if __clang_major__ >= 23
  v8i z8 = {0,0,0,0,0,0,0,0};
  __builtin_amdgcn_tensor_load_to_lds(g0, g1, z4, z4, z8, 0);
#else
  __builtin_amdgcn_tensor_load_to_lds(g0, g1, z4, z4, 0);
#endif
}

// ---------------- elementwise fp32 -> f16 ----------------
__global__ __launch_bounds__(256) void cvt_f32_f16(const float* __restrict__ in,
                                                   half_t* __restrict__ out, int n) {
  int i = blockIdx.x * 256 + threadIdx.x;
  if (i < n) out[i] = (half_t)in[i];
}

// ---------------- NT GEMM: C[M,N] f32 = A[M,K] f16 (row major) * B[N,K]^T ----------------
// workgroup: 8 waves (4x2), 128x128 macro tile; wave: 32x64 (2x4 wmma tiles, 8 wmma/K-step)
__global__ __launch_bounds__(256) void gemm_nt(const half_t* __restrict__ A,
                                               const half_t* __restrict__ B,
                                               float* __restrict__ C,
                                               int M, int N, int K) {
  const int lane = threadIdx.x & 31;
  const int wave = threadIdx.x >> 5;
  const int g  = lane >> 4;
  const int ln = lane & 15;
  const int Mb = blockIdx.y * 128 + (wave >> 1) * 32;
  const int Nb = blockIdx.x * 128 + (wave & 1) * 64;

  const v8f z = {0.f,0.f,0.f,0.f,0.f,0.f,0.f,0.f};
  v8f acc[2][4];
  #pragma unroll
  for (int i = 0; i < 2; ++i)
    #pragma unroll
    for (int j = 0; j < 4; ++j) acc[i][j] = z;

  const half_t* ap[2];
  const half_t* bp[4];
  #pragma unroll
  for (int i = 0; i < 2; ++i) ap[i] = A + (size_t)(Mb + i * 16 + ln) * K;
  #pragma unroll
  for (int j = 0; j < 4; ++j) bp[j] = B + (size_t)(Nb + j * 16 + ln) * K;

  for (int k0 = 0; k0 < K; k0 += 32) {
    v16h af[2], bf[4];
    #pragma unroll
    for (int i = 0; i < 2; ++i)
      af[i] = cat8(*(const v8h*)(ap[i] + k0 + g*8), *(const v8h*)(ap[i] + k0 + 16 + g*8));
    #pragma unroll
    for (int j = 0; j < 4; ++j)
        bf[j] = cat8(*(const v8h*)(bp[j] + k0 + g*8), *(const v8h*)(bp[j] + k0 + 16 + g*8));
    #pragma unroll
    for (int i = 0; i < 2; ++i)
      #pragma unroll
      for (int j = 0; j < 4; ++j)
        acc[i][j] = wmma16(af[i], bf[j], acc[i][j]);
    if (k0 + 256 < K) {                        // lowers to global_prefetch_b8
      __builtin_prefetch(ap[0] + k0 + 256, 0, 1);
      __builtin_prefetch(bp[0] + k0 + 256, 0, 1);
    }
  }
  #pragma unroll
  for (int i = 0; i < 2; ++i)
    #pragma unroll
    for (int r = 0; r < 8; ++r) {
      const size_t m0 = (size_t)(Mb + i * 16 + r + 8 * g);
      #pragma unroll
      for (int j = 0; j < 4; ++j)
        C[m0 * N + Nb + j * 16 + ln] = acc[i][j][r];
    }
}

// ---------------- RMS-norm + RoPE, one wave per (s, head) ----------------
__global__ __launch_bounds__(32) void normrope(const float* __restrict__ X,
                                               const float* __restrict__ cosp,
                                               const float* __restrict__ sinp,
                                               const float* __restrict__ w,
                                               half_t* __restrict__ out, int nheads) {
  const int s = blockIdx.x, h = blockIdx.y, t = threadIdx.x;
  const size_t base = ((size_t)s * nheads + h) * HDIM;
  float x0 = X[base + t],      x1 = X[base + t + 32];
  float x2 = X[base + t + 64], x3 = X[base + t + 96];
  float ss = x0*x0 + x1*x1 + x2*x2 + x3*x3;
  #pragma unroll
  for (int off = 16; off; off >>= 1) ss += __shfl_xor(ss, off, 32);
  const float r = rsqrtf(ss * (1.0f / HDIM) + RMSEPS);
  const float n0 = x0 * r * (1.f + w[t]);
  const float n1 = x1 * r * (1.f + w[t + 32]);
  const float n2 = x2 * r * (1.f + w[t + 64]);
  const float n3 = x3 * r * (1.f + w[t + 96]);
  const float* cp = cosp + (size_t)s * HDIM;
  const float* sp = sinp + (size_t)s * HDIM;
  const float o0 = n0 * cp[t]      - n2 * sp[t];
  const float o1 = n1 * cp[t + 32] - n3 * sp[t + 32];
  const float o2 = n2 * cp[t + 64] + n0 * sp[t + 64];
  const float o3 = n3 * cp[t + 96] + n1 * sp[t + 96];
  half_t* op = out + ((size_t)h * S_LEN + s) * HDIM;
  op[t]      = (half_t)o0;
  op[t + 32] = (half_t)o1;
  op[t + 64] = (half_t)o2;
  op[t + 96] = (half_t)o3;
}

// ---------------- V transpose: [S][NKV*HD] f32 -> [NKV*HD][S] f16 ----------------
__global__ __launch_bounds__(256) void vtrans(const float* __restrict__ V,
                                              half_t* __restrict__ Vt) {
  const int i = blockIdx.x * 256 + threadIdx.x;
  if (i >= S_LEN * KVD) return;
  const int s = i >> 10;
  const int e = i & (KVD - 1);
  Vt[(size_t)e * S_LEN + s] = (half_t)V[i];
}

// ---------------- flash attention: one wave per (16-query tile, head) ----------------
// K tiles (32x128 f16, 8KB) are DMA'd into LDS by the Tensor Data Mover, double
// buffered against compute via s_wait_tensorcnt. Dynamic LDS layout (base off 0):
//   [0,8192)      K tile buffer 0
//   [8192,16384)  K tile buffer 1
//   [16384,17408) P relayout buffer (16x32 f16)
__global__ __launch_bounds__(32) void attn(const half_t* __restrict__ Qr,   // [NH][S][HD]
                                           const half_t* __restrict__ Kr,   // [NKV][S][HD]
                                           const half_t* __restrict__ Vt,   // [NKV][HD][S]
                                           half_t* __restrict__ O) {        // [S][NH*HD]
  extern __shared__ half_t smem[];
  half_t* plds = smem + 2 * 32 * HDIM;

  const int qb   = blockIdx.x;
  const int h    = blockIdx.y;
  const int hkv  = h / (NHEAD / NKVH);
  const int lane = threadIdx.x;
  const int g = lane >> 4, ln = lane & 15;

  const half_t* Qh = Qr + (size_t)h   * S_LEN * HDIM;
  const half_t* Kh = Kr + (size_t)hkv * S_LEN * HDIM;
  const half_t* Vh = Vt + (size_t)hkv * HDIM * S_LEN;

  v16h qf[4];
  #pragma unroll
  for (int c = 0; c < 4; ++c)
    qf[c] = load_a_frag(Qh, HDIM, qb * 16 + ln, c * 32, g);

  const v8f z = {0.f,0.f,0.f,0.f,0.f,0.f,0.f,0.f};
  v8f acc[8];
  #pragma unroll
  for (int t = 0; t < 8; ++t) acc[t] = z;
  float mrow[8], lrow[8];
  #pragma unroll
  for (int r = 0; r < 8; ++r) { mrow[r] = -3.0e38f; lrow[r] = 0.f; }

  const int q_hi = qb * 16 + 15;
  const int nblk = (q_hi >> 5) + 1;  // KV blocks of 32 up to causal frontier

  auto compute_block = [&](int jb, const half_t* kt) {
    const int kb = jb * 32;
    v8f c0 = z, c1 = z;
    #pragma unroll
    for (int c = 0; c < 4; ++c) {
      v16h k0f = load_b_frag(kt, HDIM, ln,      c * 32, g);   // ds_load_b128 pairs
      v16h k1f = load_b_frag(kt, HDIM, 16 + ln, c * 32, g);
      c0 = wmma16(qf[c], k0f, c0);
      c1 = wmma16(qf[c], k1f, c1);
    }
    float alpha[8];
    #pragma unroll
    for (int r = 0; r < 8; ++r) {
      const int qrow = qb * 16 + r + 8 * g;
      float s0 = tanhf(c0[r] * (SCALING / SOFTCAP)) * SOFTCAP;
      float s1 = tanhf(c1[r] * (SCALING / SOFTCAP)) * SOFTCAP;
      if (kb + ln      > qrow) s0 += NEGBIG;
      if (kb + 16 + ln > qrow) s1 += NEGBIG;
      float v = fmaxf(s0, s1);
      v = fmaxf(v, __shfl_xor(v, 1, 32));
      v = fmaxf(v, __shfl_xor(v, 2, 32));
      v = fmaxf(v, __shfl_xor(v, 4, 32));
      v = fmaxf(v, __shfl_xor(v, 8, 32));
      const float mnew = fmaxf(mrow[r], v);
      alpha[r] = __expf(mrow[r] - mnew);
      const float p0 = __expf(s0 - mnew);
      const float p1 = __expf(s1 - mnew);
      c0[r] = p0; c1[r] = p1;
      float rs = p0 + p1;
      rs += __shfl_xor(rs, 1, 32);
      rs += __shfl_xor(rs, 2, 32);
      rs += __shfl_xor(rs, 4, 32);
      rs += __shfl_xor(rs, 8, 32);
      lrow[r] = lrow[r] * alpha[r] + rs;
      mrow[r] = mnew;
    }
    #pragma unroll
    for (int t = 0; t < 8; ++t)
      #pragma unroll
      for (int r = 0; r < 8; ++r) acc[t][r] *= alpha[r];

    // P: C layout -> A fragment layout via LDS
    __syncthreads();
    #pragma unroll
    for (int r = 0; r < 8; ++r) {
      plds[(r + 8*g) * 32 + ln]      = (half_t)c0[r];
      plds[(r + 8*g) * 32 + 16 + ln] = (half_t)c1[r];
    }
    __syncthreads();
    const v16h pf = cat8(*(const v8h*)(plds + ln * 32 + g * 8),
                         *(const v8h*)(plds + ln * 32 + 16 + g * 8));
    #pragma unroll
    for (int t = 0; t < 8; ++t) {
      v16h vf = load_b_frag(Vh, S_LEN, t * 16 + ln, kb, g);
      acc[t] = wmma16(pf, vf, acc[t]);
    }
  };

  // double-buffered TDM pipeline over KV blocks
  tdm_load_k_tile((uint64_t)(uintptr_t)(Kh), 0u);
  for (int jb = 0; jb < nblk - 1; ++jb) {
    const uint32_t nxt = (uint32_t)((jb + 1) & 1) * (32 * HDIM * 2);
    tdm_load_k_tile((uint64_t)(uintptr_t)(Kh + (size_t)(jb + 1) * 32 * HDIM), nxt);
    __builtin_amdgcn_s_wait_tensorcnt(1);   // tile jb landed (TDM in-order per wave)
    compute_block(jb, smem + (size_t)(jb & 1) * 32 * HDIM);
  }
  __builtin_amdgcn_s_wait_tensorcnt(0);
  compute_block(nblk - 1, smem + (size_t)((nblk - 1) & 1) * 32 * HDIM);

  float inv[8];
  #pragma unroll
  for (int r = 0; r < 8; ++r) inv[r] = 1.0f / lrow[r];
  #pragma unroll
  for (int r = 0; r < 8; ++r) {
    const size_t row = (size_t)(qb * 16 + r + 8 * g);
    #pragma unroll
    for (int t = 0; t < 8; ++t)
      O[row * HIDSZ + (size_t)h * HDIM + t * 16 + ln] = (half_t)(acc[t][r] * inv[r]);
  }
}

extern "C" void kernel_launch(void* const* d_in, const int* in_sizes, int n_in,
                              void* d_out, int out_size, void* d_ws, size_t ws_size,
                              hipStream_t stream) {
  (void)in_sizes; (void)n_in; (void)out_size; (void)ws_size;
  const float* hs   = (const float*)d_in[0];
  const float* cosp = (const float*)d_in[1];
  const float* sinp = (const float*)d_in[2];
  // d_in[3] attention_mask: causal, applied analytically in-kernel
  const float* wq   = (const float*)d_in[4];
  const float* wk   = (const float*)d_in[5];
  const float* wv   = (const float*)d_in[6];
  const float* wo   = (const float*)d_in[7];
  const float* qnw  = (const float*)d_in[8];
  const float* knw  = (const float*)d_in[9];
  float* out = (float*)d_out;

  char* ws = (char*)d_ws;
  size_t off = 0;
  half_t* hs16 = (half_t*)(ws + off); off += (size_t)S_LEN * HIDSZ * 2;   // 12.6 MB
  half_t* wq16 = (half_t*)(ws + off); off += (size_t)HIDSZ * HIDSZ * 2;  // 18.9 MB
  half_t* wk16 = (half_t*)(ws + off); off += (size_t)KVD * HIDSZ * 2;    //  6.3 MB
  half_t* wv16 = (half_t*)(ws + off); off += (size_t)KVD * HIDSZ * 2;    //  6.3 MB
  half_t* wo16 = (half_t*)(ws + off); off += (size_t)HIDSZ * HIDSZ * 2;  // 18.9 MB
  float*  q32  = (float*)(ws + off);  off += (size_t)S_LEN * HIDSZ * 4;  // 25.2 MB
  float*  k32  = (float*)(ws + off);  off += (size_t)S_LEN * KVD * 4;    //  8.4 MB
  float*  v32  = (float*)(ws + off);  off += (size_t)S_LEN * KVD * 4;    //  8.4 MB (100 MB total)
  // Safe aliases (producer launches strictly after aliased buffer's last reader; same stream):
  half_t* qr  = wq16;  // Q rotated [NH][S][HD]
  half_t* kr  = wk16;  // K rotated [NKV][S][HD]
  half_t* vt  = wv16;  // V transposed [NKV][HD][S]
  half_t* o16 = hs16;  // attention output [S][NH*HD]

  cvt_f32_f16<<<(S_LEN * HIDSZ) / 256, 256, 0, stream>>>(hs, hs16, S_LEN * HIDSZ);
  cvt_f32_f16<<<(HIDSZ * HIDSZ) / 256, 256, 0, stream>>>(wq, wq16, HIDSZ * HIDSZ);
  cvt_f32_f16<<<(KVD * HIDSZ) / 256, 256, 0, stream>>>(wk, wk16, KVD * HIDSZ);
  cvt_f32_f16<<<(KVD * HIDSZ) / 256, 256, 0, stream>>>(wv, wv16, KVD * HIDSZ);
  cvt_f32_f16<<<(HIDSZ * HIDSZ) / 256, 256, 0, stream>>>(wo, wo16, HIDSZ * HIDSZ);

  gemm_nt<<<dim3(HIDSZ / 128, S_LEN / 128), 256, 0, stream>>>(hs16, wq16, q32, S_LEN, HIDSZ, HIDSZ);
  gemm_nt<<<dim3(KVD / 128,  S_LEN / 128), 256, 0, stream>>>(hs16, wk16, k32, S_LEN, KVD, HIDSZ);
  gemm_nt<<<dim3(KVD / 128,  S_LEN / 128), 256, 0, stream>>>(hs16, wv16, v32, S_LEN, KVD, HIDSZ);

  normrope<<<dim3(S_LEN, NHEAD), 32, 0, stream>>>(q32, cosp, sinp, qnw, qr, NHEAD);
  normrope<<<dim3(S_LEN, NKVH),  32, 0, stream>>>(k32, cosp, sinp, knw, kr, NKVH);
  vtrans<<<(S_LEN * KVD) / 256, 256, 0, stream>>>(v32, vt);

  const size_t attn_lds = (2 * 32 * HDIM + 16 * 32) * sizeof(half_t);  // 17408 B
  attn<<<dim3(S_LEN / 16, NHEAD), 32, attn_lds, stream>>>(qr, kr, vt, o16);

  gemm_nt<<<dim3(HIDSZ / 128, S_LEN / 128), 256, 0, stream>>>(o16, wo16, out, S_LEN, HIDSZ, HIDSZ);
}
